// MultiHeadAttention_51960514347267
// MI455X (gfx1250) — compile-verified, matrix operand order checked by reading, not run on hardware
//
#include <hip/hip_runtime.h>
#include <cstdint>
#include <cstddef>

#define D_MODEL   1024
#define NUM_HEADS 16
#define D_HEAD    64
#define BATCH     2
#define SEQ       2048
#define M_TOTAL   (BATCH * SEQ)
#define SROW      2049   // padded LDS row stride (floats): rows land on distinct banks

typedef __attribute__((ext_vector_type(16))) __bf16 v16bf;
typedef __attribute__((ext_vector_type(8)))  float  v8f;

__device__ __forceinline__ v8f zero8() {
  v8f z = {0.f, 0.f, 0.f, 0.f, 0.f, 0.f, 0.f, 0.f};
  return z;
}

// fp32 -> bf16 round-to-nearest-even (matches hardware conversion for normals)
__device__ __forceinline__ unsigned short f2bf(float f) {
  unsigned int u = __builtin_bit_cast(unsigned int, f);
  u += 0x7fffu + ((u >> 16) & 1u);
  return (unsigned short)(u >> 16);
}

union FragBF {
  v16bf          v;
  uint4          q[2];
  unsigned short u[16];
};

// Load a 16x32 bf16 WMMA operand fragment from a row-major matrix (rows along K).
// Works for both A (row = M tile) and B (row = N tile) in NT layout.
// ISA layout: lane holds row (lane&15); K blocks offset by 8*(lane>>4).
__device__ __forceinline__ void load_frag(FragBF& f, const unsigned short* __restrict__ base,
                                          int ld, int row, int k0, int lane) {
  const unsigned short* p = base + (size_t)(row + (lane & 15)) * (size_t)ld
                                 + (size_t)(k0 + ((lane >> 4) << 3));
  f.q[0] = *(const uint4*)(p);        // j = 0..7   -> K = k0 + 8*(lane/16) + j
  f.q[1] = *(const uint4*)(p + 16);   // j = 8..15  -> K = k0 + 8*(lane/16) + 16 + (j-8)
}

__device__ __forceinline__ v8f wmma_bf16(v16bf a, v16bf b, v8f c) {
  // D = A(16x32) * B(32x16) + C, fp32 accumulate
  return __builtin_amdgcn_wmma_f32_16x16x32_bf16(false, a, false, b, (short)0, c, false, false);
}

// ---------------------------------------------------------------------------
// fp32 -> bf16 conversion
// ---------------------------------------------------------------------------
__global__ __launch_bounds__(256) void cvt_f32_bf16(const float* __restrict__ x,
                                                    unsigned short* __restrict__ y, int n) {
  int i = blockIdx.x * 256 + threadIdx.x;
  if (i < n) y[i] = f2bf(x[i]);
}

// ---------------------------------------------------------------------------
// Generic NT GEMM: C[m,n] = sum_k A[m,k] * W[n,k] + bias[n]
// A: M x K bf16 row-major, W: N x K bf16 row-major ("x @ W.T" layout)
// MODE 0: bf16 row-major out (M x N)
// MODE 1: bf16 transposed-per-head out -> vt[b][h][dh][s]
// MODE 2: fp32 row-major out (final projection)
// Workgroup = 8 waves, 128x64 output tile; wave w owns rows [m0+16w, +16),
// all 64 columns (4 accumulators). All waves share the same B fragments per
// k-step -> weight stream is served from L0/WGP$.
// ---------------------------------------------------------------------------
template <int MODE>
__global__ __launch_bounds__(256) void gemm_bf16_nt(
    const unsigned short* __restrict__ A,
    const unsigned short* __restrict__ Wt,
    const float* __restrict__ bias,
    unsigned short* __restrict__ outb,
    float* __restrict__ outf,
    int M, int N, int K) {
  const int tid  = threadIdx.x;
  const int lane = tid & 31;
  const int w    = tid >> 5;
  const int m0   = blockIdx.x * 128 + w * 16;
  const int n0   = blockIdx.y * 64;

  v8f acc[4] = {zero8(), zero8(), zero8(), zero8()};
  for (int k0 = 0; k0 < K; k0 += 32) {
    FragBF a;
    load_frag(a, A, K, m0, k0, lane);
    #pragma unroll
    for (int nt = 0; nt < 4; ++nt) {
      FragBF bb;
      load_frag(bb, Wt, K, n0 + nt * 16, k0, lane);
      acc[nt] = wmma_bf16(a.v, bb.v, acc[nt]);
    }
  }

  const int mBase = (lane >> 4) << 3;
  #pragma unroll
  for (int nt = 0; nt < 4; ++nt) {
    const int   n  = n0 + nt * 16 + (lane & 15);
    const float bn = bias[n];
    #pragma unroll
    for (int vg = 0; vg < 8; ++vg) {
      const int   m   = m0 + mBase + vg;
      const float val = acc[nt][vg] + bn;
      if (MODE == 0) {
        outb[(size_t)m * N + n] = f2bf(val);
      } else if (MODE == 1) {
        const int b = m >> 11, s = m & (SEQ - 1);
        const int h = n >> 6,  dh = n & 63;
        outb[(((size_t)b * NUM_HEADS + h) * D_HEAD + dh) * SEQ + s] = f2bf(val);
      } else {
        outf[(size_t)m * N + n] = val;
      }
    }
  }
}

// ---------------------------------------------------------------------------
// Fused attention: per (b, h, 16-query tile):
//   scores = (q . k^T) / 8 with mask  -> LDS (16 x 2048 fp32, padded)
//   softmax in LDS, stream attn to d_out (only HBM touch of the 512MB tensor)
//   ctx = attn . v via WMMA fed straight from LDS (K split across 8 waves)
// ---------------------------------------------------------------------------
__global__ __launch_bounds__(256) void attn_fused(
    const unsigned short* __restrict__ qh,   // (B,S,D) bf16
    const unsigned short* __restrict__ kh,   // (B,S,D) bf16
    const unsigned short* __restrict__ vt,   // (B,H,Dh,S) bf16
    const int* __restrict__ mask,            // (B,1,1,S)
    float* __restrict__ attnO,               // (B,H,S,S) fp32
    unsigned short* __restrict__ ctx) {      // (B,S,D) bf16
  __shared__ float smem[16 * SROW + 256 + 32];
  float* sc      = smem;               // 16 x SROW score stripe
  float* red     = smem + 16 * SROW;   // 16 x 16 reduction scratch
  float* rowstat = red + 256;          // [0..15] row max, [16..31] row sum

  const int tid  = threadIdx.x;
  const int lane = tid & 31;
  const int w    = tid >> 5;
  const int qt   = blockIdx.x;
  const int h    = blockIdx.y;
  const int b    = blockIdx.z;

  const unsigned short* qbase = qh + ((size_t)b * SEQ) * D_MODEL + h * D_HEAD;
  const unsigned short* kbase = kh + ((size_t)b * SEQ) * D_MODEL + h * D_HEAD;

  // ---- Phase 1: scores -------------------------------------------------
  FragBF aq0, aq1;
  load_frag(aq0, qbase, D_MODEL, qt * 16, 0,  lane);
  load_frag(aq1, qbase, D_MODEL, qt * 16, 32, lane);

  const int nlane = lane & 15;
  const int mHalf = (lane >> 4) << 3;
  for (int it = 0; it < 16; ++it) {
    const int ct = it * 8 + w;          // column tile 0..127
    FragBF bk0, bk1;
    load_frag(bk0, kbase, D_MODEL, ct * 16, 0,  lane);
    load_frag(bk1, kbase, D_MODEL, ct * 16, 32, lane);
    v8f c = zero8();
    c = wmma_bf16(aq0.v, bk0.v, c);
    c = wmma_bf16(aq1.v, bk1.v, c);
    const int  n  = ct * 16 + nlane;
    const bool mz = (mask[b * SEQ + n] == 0);
    #pragma unroll
    for (int vg = 0; vg < 8; ++vg) {
      const int m = vg + mHalf;
      sc[m * SROW + n] = mz ? -1e9f : (c[vg] * 0.125f);  // 1/sqrt(64)
    }
  }
  __syncthreads();

  // ---- Phase 2: softmax + attn write-out -------------------------------
  const int g = tid >> 4, cl = tid & 15;   // 16 threads per score row
  float mx = -3.4e38f;
  for (int col = cl; col < SEQ; col += 16) mx = fmaxf(mx, sc[g * SROW + col]);
  red[g * 16 + cl] = mx;
  __syncthreads();
  if (cl == 0) {
    float mm = red[g * 16];
    for (int i = 1; i < 16; ++i) mm = fmaxf(mm, red[g * 16 + i]);
    rowstat[g] = mm;
  }
  __syncthreads();
  const float rm = rowstat[g];
  float sum = 0.f;
  for (int col = cl; col < SEQ; col += 16) {
    const float e = __expf(sc[g * SROW + col] - rm);
    sc[g * SROW + col] = e;
    sum += e;
  }
  red[g * 16 + cl] = sum;
  __syncthreads();
  if (cl == 0) {
    float ss = 0.f;
    for (int i = 0; i < 16; ++i) ss += red[g * 16 + i];
    rowstat[16 + g] = ss;
  }
  __syncthreads();
  const float inv = 1.f / rowstat[16 + g];
  float* arow = attnO + (((size_t)b * NUM_HEADS + h) * SEQ + (size_t)qt * 16 + g) * SEQ;
  for (int col = cl; col < SEQ; col += 16) {
    const float p = sc[g * SROW + col] * inv;
    sc[g * SROW + col] = p;
    arow[col] = p;              // the single mandatory HBM write of attn
  }
  __syncthreads();

  // ---- Phase 3: ctx = attn . v  (K split: wave w owns K = w*256..+255) --
  v8f acc[4] = {zero8(), zero8(), zero8(), zero8()};
  const unsigned short* vtb = vt + ((size_t)b * NUM_HEADS + h) * (size_t)D_HEAD * SEQ;
  const int row = lane & 15;
  for (int ks = 0; ks < 8; ++ks) {
    const int k0 = w * 256 + ks * 32;
    FragBF a;
    #pragma unroll
    for (int j = 0; j < 16; ++j) {
      const int k = k0 + (j & 7) + mHalf + ((j >> 3) << 4);
      a.u[j] = f2bf(sc[row * SROW + k]);
    }
    #pragma unroll
    for (int nt = 0; nt < 4; ++nt) {
      FragBF bv;
      load_frag(bv, vtb, SEQ, nt * 16, k0, lane);
      acc[nt] = wmma_bf16(a.v, bv.v, acc[nt]);
    }
  }
  __syncthreads();   // all waves done reading sc

  float* part = smem;  // reuse score stripe: part[w][m][n], 8*16*64 fp32
  #pragma unroll
  for (int nt = 0; nt < 4; ++nt) {
    #pragma unroll
    for (int vg = 0; vg < 8; ++vg) {
      const int m = vg + mHalf;
      const int n = nt * 16 + row;
      part[((size_t)w * 16 + m) * 64 + n] = acc[nt][vg];
    }
  }
  __syncthreads();
  for (int e = tid; e < 1024; e += 256) {
    float s = 0.f;
    #pragma unroll
    for (int ww = 0; ww < 8; ++ww) s += part[(ww * 16 + (e >> 6)) * 64 + (e & 63)];
    const int m = e >> 6, n = e & 63;
    ctx[((size_t)b * SEQ + (size_t)qt * 16 + m) * D_MODEL + h * D_HEAD + n] = f2bf(s);
  }
}

// ---------------------------------------------------------------------------
// Host-side orchestration
// ---------------------------------------------------------------------------
extern "C" void kernel_launch(void* const* d_in, const int* in_sizes, int n_in,
                              void* d_out, int out_size, void* d_ws, size_t ws_size,
                              hipStream_t stream) {
  (void)in_sizes; (void)n_in; (void)out_size; (void)ws_size;
  const float* Q    = (const float*)d_in[0];
  const float* K    = (const float*)d_in[1];
  const float* V    = (const float*)d_in[2];
  const int*   mask = (const int*)  d_in[3];
  const float* Wq   = (const float*)d_in[4];
  const float* bq   = (const float*)d_in[5];
  const float* Wk   = (const float*)d_in[6];
  const float* bk   = (const float*)d_in[7];
  const float* Wv   = (const float*)d_in[8];
  const float* bv   = (const float*)d_in[9];
  const float* Wo   = (const float*)d_in[10];
  const float* bo   = (const float*)d_in[11];

  float* out  = (float*)d_out;                          // (B,S,D)
  float* attn = out + (size_t)M_TOTAL * D_MODEL;        // (B,H,S,S)

  char*  ws  = (char*)d_ws;
  size_t off = 0;
  auto alloc = [&](size_t bytes) -> void* {
    void* p = ws + off;
    off += (bytes + 255) & ~(size_t)255;
    return p;
  };
  const size_t nX = (size_t)M_TOTAL * D_MODEL;   // 4,194,304
  const size_t nW = (size_t)D_MODEL * D_MODEL;   // 1,048,576

  unsigned short* Qb  = (unsigned short*)alloc(nX * 2);
  unsigned short* Kb  = (unsigned short*)alloc(nX * 2);
  unsigned short* Vb  = (unsigned short*)alloc(nX * 2);
  unsigned short* Wqb = (unsigned short*)alloc(nW * 2);
  unsigned short* Wkb = (unsigned short*)alloc(nW * 2);
  unsigned short* Wvb = (unsigned short*)alloc(nW * 2);
  unsigned short* Wob = (unsigned short*)alloc(nW * 2);
  unsigned short* qhB = (unsigned short*)alloc(nX * 2);  // (B,S,D)
  unsigned short* khB = (unsigned short*)alloc(nX * 2);  // (B,S,D)
  unsigned short* vtB = (unsigned short*)alloc(nX * 2);  // (B,H,Dh,S)
  unsigned short* ctB = (unsigned short*)alloc(nX * 2);  // (B,S,D)

  const dim3 blk(256);

  // 1) fp32 -> bf16 conversions
  cvt_f32_bf16<<<(unsigned)(nX / 256), blk, 0, stream>>>(Q,  Qb,  (int)nX);
  cvt_f32_bf16<<<(unsigned)(nX / 256), blk, 0, stream>>>(K,  Kb,  (int)nX);
  cvt_f32_bf16<<<(unsigned)(nX / 256), blk, 0, stream>>>(V,  Vb,  (int)nX);
  cvt_f32_bf16<<<(unsigned)(nW / 256), blk, 0, stream>>>(Wq, Wqb, (int)nW);
  cvt_f32_bf16<<<(unsigned)(nW / 256), blk, 0, stream>>>(Wk, Wkb, (int)nW);
  cvt_f32_bf16<<<(unsigned)(nW / 256), blk, 0, stream>>>(Wv, Wvb, (int)nW);
  cvt_f32_bf16<<<(unsigned)(nW / 256), blk, 0, stream>>>(Wo, Wob, (int)nW);

  // 2) projections (x @ W.T + b)
  const dim3 ggrid(M_TOTAL / 128, D_MODEL / 64);
  gemm_bf16_nt<0><<<ggrid, blk, 0, stream>>>(Qb, Wqb, bq, qhB, nullptr, M_TOTAL, D_MODEL, D_MODEL);
  gemm_bf16_nt<0><<<ggrid, blk, 0, stream>>>(Kb, Wkb, bk, khB, nullptr, M_TOTAL, D_MODEL, D_MODEL);
  gemm_bf16_nt<1><<<ggrid, blk, 0, stream>>>(Vb, Wvb, bv, vtB, nullptr, M_TOTAL, D_MODEL, D_MODEL);

  // 3) fused scores + softmax + attn write + attn.V
  const dim3 agrid(SEQ / 16, NUM_HEADS, BATCH);
  attn_fused<<<agrid, blk, 0, stream>>>(qhB, khB, vtB, mask, attn, ctB);

  // 4) output projection
  gemm_bf16_nt<2><<<ggrid, blk, 0, stream>>>(ctB, Wob, bo, nullptr, out, M_TOTAL, D_MODEL, D_MODEL);
}
